// CustomTaylorLayer_45243185496818
// MI455X (gfx1250) — matrix operand order repaired and checked by reading, not compiled, not run on hardware
//
#include <hip/hip_runtime.h>
#include <hip/hip_bf16.h>
#include <math.h>
#include <stdint.h>

typedef __attribute__((ext_vector_type(16))) _Float16 v16h;
typedef __attribute__((ext_vector_type(8)))  _Float16 v8h;
typedef __attribute__((ext_vector_type(2)))  _Float16 h2;
typedef __attribute__((ext_vector_type(8)))  float    v8f;
typedef __attribute__((ext_vector_type(4)))  float    v4f;

#define B_DIM   8192
#define IN_DIM  1024
#define OUT_DIM 1024
#define ORDER_P1 8
#define K_DIM   (IN_DIM * ORDER_P1)   // 8192

// ---------------------------------------------------------------------------
// Branch-free tanh: native V_TANH_F32 if the builtin exists, else exp2+rcp.
// ---------------------------------------------------------------------------
__device__ __forceinline__ float fast_tanh(float x) {
#if __has_builtin(__builtin_amdgcn_tanhf)
    return __builtin_amdgcn_tanhf(x);
#else
    float e = __builtin_amdgcn_exp2f(x * 2.885390081777927f); // 2*log2(e)
    return 1.0f - 2.0f * __builtin_amdgcn_rcpf(e + 1.0f);
#endif
}

__device__ __forceinline__ void wait_asynccnt0() {
#if __has_builtin(__builtin_amdgcn_s_wait_asynccnt)
    __builtin_amdgcn_s_wait_asynccnt(0);
#else
    asm volatile("s_wait_asynccnt 0x0" ::: "memory");
#endif
}

// ---------------------------------------------------------------------------
// Kernel 0: convert coef f32 [OUT, IN, 8] -> f16 flat [OUT, K] in workspace.
// ---------------------------------------------------------------------------
__global__ void coef_to_f16_kernel(const float* __restrict__ c,
                                   _Float16* __restrict__ h, int nvec) {
    int i      = blockIdx.x * blockDim.x + threadIdx.x;
    int stride = gridDim.x * blockDim.x;
    for (; i < nvec; i += stride) {
        v4f a = ((const v4f*)c)[2 * i];
        v4f b = ((const v4f*)c)[2 * i + 1];
        v8h o;
        o[0] = (_Float16)a.x; o[1] = (_Float16)a.y;
        o[2] = (_Float16)a.z; o[3] = (_Float16)a.w;
        o[4] = (_Float16)b.x; o[5] = (_Float16)b.y;
        o[6] = (_Float16)b.z; o[7] = (_Float16)b.w;
        ((v8h*)h)[i] = o;
    }
}

// ---------------------------------------------------------------------------
// Kernel 1: fused tanh-polynomial + WMMA GEMM.
// Block: 256 threads (8 waves). Output tile 256(M) x 128(N), K tile 64.
// Waves in 4x2 grid; each wave owns a 64x64 sub-tile (4x4 WMMA fragments)
// so LDS reads per WMMA are halved vs a 32x64 wave tile (1:1 ds:wmma).
// Double-buffered LDS; coef tile staged with async global->LDS.
// ---------------------------------------------------------------------------
__global__ __launch_bounds__(256)
void taylor_wmma_gemm_kernel(const float* __restrict__ x,
                             const float* __restrict__ trange,
                             const _Float16* __restrict__ coefh,
                             float* __restrict__ out) {
    __shared__ __align__(16) _Float16 As[2][256][72];  // 72 KB
    __shared__ __align__(16) _Float16 Bs[2][128][72];  // 36 KB

    const int tid  = threadIdx.x;
    const int lane = tid & 31;
    const int wave = tid >> 5;        // 0..7
    const int wm   = wave >> 1;       // 0..3  -> M offset wm*64
    const int wn   = wave & 1;        // 0..1  -> N offset wn*64
    const int row0 = blockIdx.y * 256;
    const int col0 = blockIdx.x * 128;
    const float tr = trange[0];

    const int lhalf = lane & 15;
    const int lhi   = lane >> 4;

    v8f zero = {};
    v8f acc[4][4];
    #pragma unroll
    for (int mi = 0; mi < 4; ++mi)
        #pragma unroll
        for (int ni = 0; ni < 4; ++ni)
            acc[mi][ni] = zero;

    // ---- staging helpers (uniform control flow; all lanes active) --------
    auto stage_A = [&](int buf, int kt) {
        const int i0 = kt >> 3;
        #pragma unroll
        for (int e = 0; e < 8; ++e) {
            int idx = tid + e * 256;          // 0..2047
            int m   = idx >> 3;               // 0..255
            int ic  = idx & 7;
            float xv = x[(size_t)(row0 + m) * IN_DIM + i0 + ic];
            float t  = fast_tanh(xv * tr);
            _Float16 th = (_Float16)t;
            _Float16 t2 = th * th;
            h2 tt; tt[0] = t2; tt[1] = t2;
            union { v8h v; h2 p[4]; } u;
            u.p[0][0] = (_Float16)1.0f; u.p[0][1] = th;   // t^0, t^1
            u.p[1] = u.p[0] * tt;                         // t^2, t^3
            u.p[2] = u.p[1] * tt;                         // t^4, t^5
            u.p[3] = u.p[2] * tt;                         // t^6, t^7
            *(v8h*)&As[buf][m][ic * 8] = u.v;
        }
    };

    auto stage_B_async = [&](int buf, int kt) {
        #pragma unroll
        for (int e = 0; e < 4; ++e) {
            int idx = tid + e * 256;          // 0..1023
            int n   = idx >> 3;               // 0..127
            int kc  = idx & 7;
            uint64_t g = (uint64_t)(uintptr_t)
                &coefh[(size_t)(col0 + n) * K_DIM + kt + kc * 8];
            uint32_t l = (uint32_t)(uintptr_t)&Bs[buf][n][kc * 8];
            asm volatile("global_load_async_to_lds_b128 %0, %1, off"
                         :: "v"(l), "v"(g) : "memory");
        }
    };

    // ---- prologue: stage tile 0 into buffer 0 ----------------------------
    stage_B_async(0, 0);
    stage_A(0, 0);
    wait_asynccnt0();
    __syncthreads();

    for (int kt = 0; kt < K_DIM; kt += 64) {
        const int cur = (kt >> 6) & 1;
        const int nxt = cur ^ 1;

        if (kt + 64 < K_DIM) {
            stage_B_async(nxt, kt + 64);
            stage_A(nxt, kt + 64);
        }

        // ---- two 16x16x32 WMMA chunks over current K tile ----------------
        #pragma unroll
        for (int kk = 0; kk < 64; kk += 32) {
            // A frags: lane<16 -> K[kk..+7] & K[kk+16..+23]; lane>=16 -> +8
            v16h afr[4];
            #pragma unroll
            for (int mi = 0; mi < 4; ++mi) {
                int r  = wm * 64 + mi * 16 + lhalf;
                int kb = kk + (lhi << 3);
                union { v16h v; v8h h[2]; } u;
                u.h[0] = *(const v8h*)&As[cur][r][kb];
                u.h[1] = *(const v8h*)&As[cur][r][kb + 16];
                afr[mi] = u.v;
            }
            // B frags: lane<16 -> N=lane, K[kk..+7] & K[kk+16..+23]; lane>=16 -> +8
            v16h bfr[4];
            #pragma unroll
            for (int ni = 0; ni < 4; ++ni) {
                int n  = wn * 64 + ni * 16 + lhalf;
                int kb = kk + (lhi << 3);
                union { v16h v; v8h h[2]; } u;
                u.h[0] = *(const v8h*)&Bs[cur][n][kb];
                u.h[1] = *(const v8h*)&Bs[cur][n][kb + 16];
                bfr[ni] = u.v;
            }
            #pragma unroll
            for (int mi = 0; mi < 4; ++mi)
                #pragma unroll
                for (int ni = 0; ni < 4; ++ni)
                    acc[mi][ni] = __builtin_amdgcn_wmma_f32_16x16x32_f16(
                        false, afr[mi], false, bfr[ni],
                        (short)0, acc[mi][ni], false, false);
        }

        wait_asynccnt0();
        __syncthreads();
    }

    // ---- store (nontemporal: keep coef/x resident in L2) -----------------
    // D layout: VGPR v -> M = v + (lane>>4)*8, N = lane&15
    #pragma unroll
    for (int mi = 0; mi < 4; ++mi) {
        #pragma unroll
        for (int ni = 0; ni < 4; ++ni) {
            int colg  = col0 + wn * 64 + ni * 16 + lhalf;
            int rbase = row0 + wm * 64 + mi * 16 + (lhi << 3);
            #pragma unroll
            for (int v = 0; v < 8; ++v)
                __builtin_nontemporal_store(
                    acc[mi][ni][v],
                    &out[(size_t)(rbase + v) * OUT_DIM + colg]);
        }
    }
}

// ---------------------------------------------------------------------------
extern "C" void kernel_launch(void* const* d_in, const int* in_sizes, int n_in,
                              void* d_out, int out_size, void* d_ws, size_t ws_size,
                              hipStream_t stream) {
    const float* x    = (const float*)d_in[0];
    const float* tr   = (const float*)d_in[1];
    const float* coef = (const float*)d_in[2];
    float* out        = (float*)d_out;
    _Float16* coefh   = (_Float16*)d_ws;   // 16 MB of workspace

    int nvec = (OUT_DIM * K_DIM) / 8;
    coef_to_f16_kernel<<<2048, 256, 0, stream>>>(coef, coefh, nvec);

    dim3 grid(OUT_DIM / 128, B_DIM / 256); // (8, 32)
    taylor_wmma_gemm_kernel<<<grid, 256, 0, stream>>>(x, tr, coefh, out);
}
